// GTrXLLayer_1340029796448
// MI455X (gfx1250) — matrix operand order repaired and checked by reading, not compiled, not run on hardware
//
#include <hip/hip_runtime.h>
#include <hip/hip_bf16.h>

// ---------------------------------------------------------------------------
// GTrXL layer forward for MI455X (gfx1250, wave32, WMMA + async/TDM copies).
// All big GEMMs: bf16 x bf16 -> f32 via v_wmma_f32_16x16x32_bf16, with
// double-buffered LDS staged by GLOBAL_LOAD_ASYNC_TO_LDS_B128 (ASYNCcnt).
// Attention q-tiles staged by TENSOR_LOAD_TO_LDS (TDM, TENSORcnt).
// ---------------------------------------------------------------------------

#define S_LEN   256
#define BATCH   8
#define DMODEL  1024
#define NHEAD   16
#define HDIM    64
#define MEMLEN  256
#define KLEN    512
#define DFF_    4096
#define SB      2048   // S*B rows
#define KB      4096   // K*B rows

typedef __attribute__((ext_vector_type(16))) __bf16 v16bf;
typedef __attribute__((ext_vector_type(8)))  __bf16 v8bf;
typedef __attribute__((ext_vector_type(8)))  float  v8f;
typedef __attribute__((ext_vector_type(4)))  unsigned u32x4;
typedef __attribute__((ext_vector_type(8)))  unsigned u32x8;

// ---- WMMA fragment helpers (16x16x32 bf16; A and B both use contiguous k = 16*half + s) ------
// A (16x32, MxK): lane<16 -> row=lane, K 0..15 contiguous ; lane>=16 -> K 16..31 contiguous
__device__ __forceinline__ v16bf load_a_frag(const __hip_bfloat16* tile, int ld,
                                             int row16, int half, int kOff) {
  union { v16bf v; v8bf h[2]; } u;
  const __hip_bfloat16* p = tile + (size_t)row16 * ld + kOff;
  u.h[0] = *(const v8bf*)(p + half * 16);
  u.h[1] = *(const v8bf*)(p + half * 16 + 8);
  return u.v;
}
// B (32x16, KxN) fed from an N-major (transposed) tile: lane<16 -> col=lane,
// K 0..15 contiguous; lane>=16 -> col=lane-16, K 16..31 contiguous.
__device__ __forceinline__ v16bf load_bt_frag(const __hip_bfloat16* tileT, int ld,
                                              int row16, int half, int kOff) {
  return *(const v16bf*)(tileT + (size_t)row16 * ld + kOff + half * 16);
}
__device__ __forceinline__ v8f wmma_bf16(v16bf a, v16bf b, v8f c) {
  return __builtin_amdgcn_wmma_f32_16x16x32_bf16(false, a, false, b, (short)0, c,
                                                 false, false);
}

// ---- CDNA5 async global->LDS copy (ISA ch.10 / CDNA5 ISA) -----
// Per-lane 16B copy, tracked with ASYNCcnt. Flat-address low 32 bits are the
// LDS byte offset (aperture truncation, ISA 10.2).
__device__ __forceinline__ void async_copy_b128(void* lds_dst, const void* gsrc) {
  unsigned ldsOff = (unsigned)(size_t)lds_dst;
  asm volatile("global_load_async_to_lds_b128 %0, %1, off"
               :: "v"(ldsOff), "v"(gsrc) : "memory");
}
__device__ __forceinline__ void wait_asynccnt0() {
  asm volatile("s_wait_asynccnt 0x0" ::: "memory");
}

// ---- CDNA5 Tensor Data Mover 2D tile load (ISA §7/§8, D# groups 0..3) -----
// data_size=2B elements, tile tileW x tileH, row stride in elements.
__device__ __forceinline__ void tdm_load_2d(void* lds_dst, const void* gsrc,
                                            unsigned tileW, unsigned tileH,
                                            unsigned long long rowStrideElems) {
  unsigned long long ga = (unsigned long long)(size_t)gsrc;
  u32x4 g0; u32x8 g1; u32x4 gz = {0u, 0u, 0u, 0u};
  g0.x = 1u;                                   // count=1 (valid user D#)
  g0.y = (unsigned)(size_t)lds_dst;            // lds_addr (byte offset)
  g0.z = (unsigned)(ga & 0xFFFFFFFFu);         // global_addr[31:0]
  g0.w = (unsigned)((ga >> 32) & 0x01FFFFFFu)  // global_addr[56:32]
         | 0x80000000u;                        // type=2 ("image")
  g1[0] = 0x00010000u;                         // data_size=1 -> 2-byte elems
  g1[1] = (tileW & 0xFFFFu) << 16;             // tensor_dim0[15:0] @63:48
  g1[2] = (tileW >> 16) | ((tileH & 0xFFFFu) << 16);  // dim0 hi | tensor_dim1 lo
  g1[3] = (tileH >> 16) | ((tileW & 0xFFFFu) << 16);  // dim1 hi | tile_dim0
  g1[4] = tileH & 0xFFFFu;                     // tile_dim1 (tile_dim2=0)
  g1[5] = (unsigned)(rowStrideElems & 0xFFFFFFFFu);   // tensor_dim0_stride lo
  g1[6] = (unsigned)((rowStrideElems >> 32) & 0xFFFFu);
  g1[7] = 0u;
  asm volatile("tensor_load_to_lds %0, %1, %2, %3"
               :: "s"(g0), "s"(g1), "s"(gz), "s"(gz) : "memory");
}

// ---------------------------------------------------------------------------
// GEMM: C(M,N) = A(M,K) @ Wt^T, A row-major bf16 (M x K), Wt is N x K
// (pre-transposed weight) so both LDS tiles are row-chunk async copies.
// Block tile 128x128, 8 waves, wave tile 32x64 (2x4 WMMA accumulators),
// double-buffered LDS staged with async copies (one barrier per k-step).
// flags: 1=accumulate into Cf, 2=add bias[col], 4=relu. Cf/Cb optional outs.
// ---------------------------------------------------------------------------
#define GBM 128
#define GBN 128
#define GBK 32

__launch_bounds__(256)
__global__ void gemm_kernel(const __hip_bfloat16* __restrict__ A,
                            const __hip_bfloat16* __restrict__ Wt,  // N x K
                            int M, int N, int Kd,
                            float* __restrict__ Cf,
                            __hip_bfloat16* __restrict__ Cb,
                            const float* __restrict__ bias,
                            int flags) {
  __shared__ __align__(128) __hip_bfloat16 sA[2][GBM][GBK];  // 2 x 8 KB
  __shared__ __align__(128) __hip_bfloat16 sW[2][GBN][GBK];  // 2 x 8 KB
  const int t = threadIdx.x;
  const int wid = t >> 5;
  const int lane = t & 31;
  const int half = lane >> 4;
  const int row16 = lane & 15;
  const int bm = blockIdx.x * GBM;
  const int bn = blockIdx.y * GBN;
  const int waveM = (wid & 3) * 32;
  const int waveN = (wid >> 2) * 64;

  v8f acc[2][4] = {};

  const int srow = t >> 1;            // 0..127
  const int sseg = (t & 1) * 16;      // 0 or 16 (elements)

  const __hip_bfloat16* Abase = A + (size_t)(bm + srow) * Kd + sseg;
  const __hip_bfloat16* Wbase = Wt + (size_t)(bn + srow) * Kd + sseg;

  // Prologue: stage k-chunk 0 into buffer 0 with async copies.
  async_copy_b128(&sA[0][srow][sseg], Abase);
  async_copy_b128(&sA[0][srow][sseg + 8], Abase + 8);
  async_copy_b128(&sW[0][srow][sseg], Wbase);
  async_copy_b128(&sW[0][srow][sseg + 8], Wbase + 8);
  wait_asynccnt0();
  __syncthreads();

  int buf = 0;
  for (int k0 = 0; k0 < Kd; k0 += GBK) {
    // Kick off async staging of the next k-chunk into the other buffer.
    if (k0 + GBK < Kd) {
      const __hip_bfloat16* an = Abase + k0 + GBK;
      const __hip_bfloat16* wn = Wbase + k0 + GBK;
      async_copy_b128(&sA[buf ^ 1][srow][sseg], an);
      async_copy_b128(&sA[buf ^ 1][srow][sseg + 8], an + 8);
      async_copy_b128(&sW[buf ^ 1][srow][sseg], wn);
      async_copy_b128(&sW[buf ^ 1][srow][sseg + 8], wn + 8);
    }
    // Consume current buffer: 8 WMMAs per wave.
    v16bf af0 = load_a_frag(&sA[buf][waveM][0],      GBK, row16, half, 0);
    v16bf af1 = load_a_frag(&sA[buf][waveM + 16][0], GBK, row16, half, 0);
    #pragma unroll
    for (int ni = 0; ni < 4; ++ni) {
      v16bf bfg = load_bt_frag(&sW[buf][waveN + ni * 16][0], GBK, row16, half, 0);
      acc[0][ni] = wmma_bf16(af0, bfg, acc[0][ni]);
      acc[1][ni] = wmma_bf16(af1, bfg, acc[1][ni]);
    }
    wait_asynccnt0();   // next buffer's data landed (this wave's copies)
    __syncthreads();    // ... and every other wave's copies too
    buf ^= 1;
  }

  // Epilogue. C/D layout: VGPR r -> M = r + half*8, N = row16 (ISA 7.12.2).
  #pragma unroll
  for (int mi = 0; mi < 2; ++mi)
    #pragma unroll
    for (int ni = 0; ni < 4; ++ni)
      #pragma unroll
      for (int r = 0; r < 8; ++r) {
        int row = bm + waveM + mi * 16 + r + half * 8;
        int col = bn + waveN + ni * 16 + row16;
        size_t idx = (size_t)row * N + col;
        float v = acc[mi][ni][r];
        if (flags & 1) v += Cf[idx];
        if (flags & 2) v += bias[col];
        if (flags & 4) v = fmaxf(v, 0.0f);
        if (Cf) Cf[idx] = v;
        if (Cb) Cb[idx] = __float2bfloat16(v);
      }
}

// ---------------------------------------------------------------------------
// Relative-position attention. One workgroup = (b, h, 32-query tile).
// scores = (q+u)kT + gather((q+v)rT) ; mask ; softmax ; y = P @ V.
// q-tiles staged via TDM tensor_load_to_lds; 136 KB dynamic LDS (CDNA5
// allows up to 320 KB per workgroup).
// ---------------------------------------------------------------------------
__launch_bounds__(256)
__global__ void attn_kernel(const __hip_bfloat16* __restrict__ qu,
                            const __hip_bfloat16* __restrict__ qv,
                            const __hip_bfloat16* __restrict__ kbuf,
                            const __hip_bfloat16* __restrict__ vbuf,
                            const __hip_bfloat16* __restrict__ rbuf,
                            float* __restrict__ ybuf) {
  extern __shared__ __align__(128) char smem[];
  __hip_bfloat16* sQu = (__hip_bfloat16*)smem;                    //  4 KB
  __hip_bfloat16* sQv = (__hip_bfloat16*)(smem + 4096);           //  4 KB
  float* sScore = (float*)(smem + 8192);                          // 64 KB
  float* sQr    = (float*)(smem + 8192 + 65536);                  // 64 KB
  __hip_bfloat16* sP  = (__hip_bfloat16*)sQr;     // overlay (qr consumed)
  __hip_bfloat16* sVT = (__hip_bfloat16*)sScore;  // overlay (scores consumed)

  const int t = threadIdx.x, wid = t >> 5, lane = t & 31;
  const int half = lane >> 4, row16 = lane & 15;
  const int q0 = blockIdx.x * 32;
  const int h  = blockIdx.y;
  const int b  = blockIdx.z;

  // Stage 32x64 q tiles (u/v biases pre-folded) with the Tensor Data Mover.
  // Tile row r corresponds to query s=q0+r at global row (s*B+b), cols h*64..
  if (wid == 0) {
    const size_t gelem = ((size_t)q0 * BATCH + b) * DMODEL + h * HDIM;
    tdm_load_2d(sQu, qu + gelem, HDIM, 32, (unsigned long long)BATCH * DMODEL);
    tdm_load_2d(sQv, qv + gelem, HDIM, 32, (unsigned long long)BATCH * DMODEL);
    __builtin_amdgcn_s_wait_tensorcnt(0);
  }
  __syncthreads();

  // Content scores (q_u @ kT) and raw position scores (q_v @ rT), WMMA.
  // 64 output tiles (2 q-subtiles x 32 key-tiles) split across 8 waves.
  for (int tile = wid; tile < 64; tile += 8) {
    int mi = tile & 1;
    int kt = tile >> 1;
    v8f acc = {}, accr = {};
    #pragma unroll
    for (int ks = 0; ks < 2; ++ks) {
      v16bf a  = load_a_frag(sQu + mi * 16 * HDIM, HDIM, row16, half, ks * 32);
      v16bf av = load_a_frag(sQv + mi * 16 * HDIM, HDIM, row16, half, ks * 32);
      int key = kt * 16 + row16;  // lane picks its key column; HD contiguous
      v16bf bk = *(const v16bf*)(kbuf + ((size_t)key * BATCH + b) * DMODEL +
                                 h * HDIM + ks * 32 + half * 16);
      v16bf br = *(const v16bf*)(rbuf + (size_t)key * DMODEL +
                                 h * HDIM + ks * 32 + half * 16);
      acc  = wmma_bf16(a,  bk, acc);
      accr = wmma_bf16(av, br, accr);
    }
    #pragma unroll
    for (int r = 0; r < 8; ++r) {
      int ql = mi * 16 + r + half * 8;
      int kk = kt * 16 + row16;
      sScore[ql * KLEN + kk] = acc[r];
      sQr[ql * KLEN + kk]    = accr[r];
    }
  }
  __syncthreads();

  // Rel-shift gather + causal mask + scale (1/sqrt(64)).
  for (int e = t; e < 32 * KLEN; e += 256) {
    int ql = e >> 9, k = e & (KLEN - 1);
    int sabs = q0 + ql;
    int ridx = sabs + MEMLEN - k;            // clip(q_pos - k_pos, 0, K-1)
    float v;
    if (k > sabs + MEMLEN) v = -1e30f;       // masked
    else                   v = (sScore[e] + sQr[ql * KLEN + ridx]) * 0.125f;
    sScore[e] = v;
  }
  __syncthreads();

  // Softmax per row: 8 waves x 4 rows; write bf16 probs into sP.
  for (int rr = 0; rr < 4; ++rr) {
    int ql = wid * 4 + rr;
    float vals[16];
    float m = -1e30f;
    #pragma unroll
    for (int j = 0; j < 16; ++j) {
      vals[j] = sScore[ql * KLEN + lane + j * 32];
      m = fmaxf(m, vals[j]);
    }
    #pragma unroll
    for (int o = 16; o > 0; o >>= 1) m = fmaxf(m, __shfl_xor(m, o, 32));
    float sum = 0.f;
    #pragma unroll
    for (int j = 0; j < 16; ++j) { vals[j] = __expf(vals[j] - m); sum += vals[j]; }
    #pragma unroll
    for (int o = 16; o > 0; o >>= 1) sum += __shfl_xor(sum, o, 32);
    float inv = 1.0f / sum;
    #pragma unroll
    for (int j = 0; j < 16; ++j)
      sP[ql * KLEN + lane + j * 32] = __float2bfloat16(vals[j] * inv);
  }
  __syncthreads();

  // Stage V transposed: sVT[n][kk] (overwrites score region).
  for (int e = t; e < HDIM * KLEN; e += 256) {
    int kk = e >> 6, n = e & 63;
    sVT[n * KLEN + kk] = vbuf[((size_t)kk * BATCH + b) * DMODEL + h * HDIM + n];
  }
  __syncthreads();

  // y(32x64) = P(32x512) @ V(512x64): 8 output tiles, one per wave.
  {
    int mi = wid & 1, ni = wid >> 1;
    v8f acc = {};
    for (int ks = 0; ks < 16; ++ks) {
      v16bf a  = load_a_frag(sP + mi * 16 * KLEN, KLEN, row16, half, ks * 32);
      v16bf bv = load_bt_frag(sVT + ni * 16 * KLEN, KLEN, row16, half, ks * 32);
      acc = wmma_bf16(a, bv, acc);
    }
    #pragma unroll
    for (int r = 0; r < 8; ++r) {
      int srow = q0 + mi * 16 + r + half * 8;
      int col = h * HDIM + ni * 16 + row16;
      ybuf[((size_t)srow * BATCH + b) * DMODEL + col] = acc[r];
    }
  }
}

// ---------------------------------------------------------------------------
// LayerNorm over D=1024: one block per row, bf16 output.
// ---------------------------------------------------------------------------
__launch_bounds__(256)
__global__ void ln_kernel(const float* __restrict__ src,
                          const float* __restrict__ g,
                          const float* __restrict__ bb,
                          __hip_bfloat16* __restrict__ dst) {
  int row = blockIdx.x, t = threadIdx.x;
  const float* p = src + (size_t)row * DMODEL;
  float vals[4], s = 0.f, ss = 0.f;
  #pragma unroll
  for (int i = 0; i < 4; ++i) {
    float v = p[t + i * 256];
    vals[i] = v; s += v; ss += v * v;
  }
  #pragma unroll
  for (int o = 16; o > 0; o >>= 1) {
    s  += __shfl_xor(s, o, 32);
    ss += __shfl_xor(ss, o, 32);
  }
  __shared__ float ws[8], wss[8];
  if ((t & 31) == 0) { ws[t >> 5] = s; wss[t >> 5] = ss; }
  __syncthreads();
  float S = 0.f, SS = 0.f;
  #pragma unroll
  for (int w = 0; w < 8; ++w) { S += ws[w]; SS += wss[w]; }
  float mean = S * (1.0f / DMODEL);
  float var  = SS * (1.0f / DMODEL) - mean * mean;
  float inv  = rsqrtf(var + 1e-5f);
  #pragma unroll
  for (int i = 0; i < 4; ++i) {
    int c = t + i * 256;
    dst[(size_t)row * DMODEL + c] =
        __float2bfloat16((vals[i] - mean) * inv * g[c] + bb[c]);
  }
}

// ---------------------------------------------------------------------------
// Elementwise kernels (grid-stride).
// ---------------------------------------------------------------------------
__global__ void cvt_kernel(const float* __restrict__ in,
                           __hip_bfloat16* __restrict__ out, long long n) {
  long long i = (long long)blockIdx.x * 256 + threadIdx.x;
  long long stride = (long long)gridDim.x * 256;
  for (; i < n; i += stride) out[i] = __float2bfloat16(in[i]);
}
// f32 (K x N) -> bf16 transposed (N x K) for async-friendly GEMM staging.
__global__ void cvtT_kernel(const float* __restrict__ in,
                            __hip_bfloat16* __restrict__ out,
                            int Kd, int N) {
  long long total = (long long)Kd * N;
  long long i = (long long)blockIdx.x * 256 + threadIdx.x;
  long long stride = (long long)gridDim.x * 256;
  for (; i < total; i += stride) {
    int k = (int)(i / N), n = (int)(i - (long long)k * N);
    out[(size_t)n * Kd + k] = __float2bfloat16(in[i]);
  }
}
__global__ void qbias_kernel(const float* __restrict__ q,
                             const float* __restrict__ u,
                             const float* __restrict__ v,
                             __hip_bfloat16* __restrict__ qu,
                             __hip_bfloat16* __restrict__ qv, long long n) {
  long long i = (long long)blockIdx.x * 256 + threadIdx.x;
  long long stride = (long long)gridDim.x * 256;
  for (; i < n; i += stride) {
    int col = (int)(i & (DMODEL - 1));  // col == h*HD + hd, matches (H,HD) flat
    float qq = q[i];
    qu[i] = __float2bfloat16(qq + u[col]);
    qv[i] = __float2bfloat16(qq + v[col]);
  }
}
__global__ void sigmoid_kernel(float* __restrict__ tio,
                               const float* __restrict__ bias, long long n) {
  long long i = (long long)blockIdx.x * 256 + threadIdx.x;
  long long stride = (long long)gridDim.x * 256;
  for (; i < n; i += stride) {
    float x = tio[i];
    if (bias) x -= bias[i & (DMODEL - 1)];
    tio[i] = 1.0f / (1.0f + __expf(-x));
  }
}
__global__ void tanh_kernel(float* __restrict__ tio, long long n) {
  long long i = (long long)blockIdx.x * 256 + threadIdx.x;
  long long stride = (long long)gridDim.x * 256;
  for (; i < n; i += stride) tio[i] = tanhf(tio[i]);
}
__global__ void relu_bf_kernel(const float* __restrict__ in,
                               __hip_bfloat16* __restrict__ out, long long n) {
  long long i = (long long)blockIdx.x * 256 + threadIdx.x;
  long long stride = (long long)gridDim.x * 256;
  for (; i < n; i += stride) out[i] = __float2bfloat16(fmaxf(in[i], 0.0f));
}
__global__ void mul_bf_kernel(const float* __restrict__ a,
                              const float* __restrict__ b,
                              __hip_bfloat16* __restrict__ out, long long n) {
  long long i = (long long)blockIdx.x * 256 + threadIdx.x;
  long long stride = (long long)gridDim.x * 256;
  for (; i < n; i += stride) out[i] = __float2bfloat16(a[i] * b[i]);
}
__global__ void gru_out_kernel(const float* __restrict__ z,
                               const float* __restrict__ x,
                               const float* __restrict__ hh,
                               float* __restrict__ out, long long n) {
  long long i = (long long)blockIdx.x * 256 + threadIdx.x;
  long long stride = (long long)gridDim.x * 256;
  for (; i < n; i += stride) {
    float zz = z[i];
    out[i] = (1.0f - zz) * x[i] + zz * hh[i];
  }
}

// ---------------------------------------------------------------------------
// Host driver.
// ---------------------------------------------------------------------------
extern "C" void kernel_launch(void* const* d_in, const int* in_sizes, int n_in,
                              void* d_out, int out_size, void* d_ws, size_t ws_size,
                              hipStream_t stream) {
  (void)in_sizes; (void)n_in; (void)out_size; (void)ws_size;

  const float* x      = (const float*)d_in[0];
  const float* memory = (const float*)d_in[1];
  const float* relpos = (const float*)d_in[2];
  // d_in[3] attn_mask: deterministic causal band, recomputed in-kernel.
  const float* ln1_g = (const float*)d_in[4];
  const float* ln1_b = (const float*)d_in[5];
  const float* ln2_g = (const float*)d_in[6];
  const float* ln2_b = (const float*)d_in[7];
  const float* Wq  = (const float*)d_in[8];
  const float* Wk  = (const float*)d_in[9];
  const float* Wv  = (const float*)d_in[10];
  const float* Wrp = (const float*)d_in[11];
  const float* Wo  = (const float*)d_in[12];
  const float* u_bias = (const float*)d_in[13];
  const float* v_bias = (const float*)d_in[14];
  const float* ff_w1 = (const float*)d_in[15];
  const float* ff_b1 = (const float*)d_in[16];
  const float* ff_w2 = (const float*)d_in[17];
  const float* ff_b2 = (const float*)d_in[18];
  const float* gaW[6]; const float* gfW[6];
  for (int i = 0; i < 6; ++i) gaW[i] = (const float*)d_in[19 + i];
  const float* ga_bias = (const float*)d_in[25];
  for (int i = 0; i < 6; ++i) gfW[i] = (const float*)d_in[26 + i];
  const float* gf_bias = (const float*)d_in[32];

  char* base = (char*)d_ws;
  size_t off = 0;
  auto alloc = [&](size_t bytes) -> void* {
    void* p = base + off;
    off = (off + bytes + 255) & ~(size_t)255;
    return p;
  };
  typedef __hip_bfloat16 bf;
  const size_t DD = (size_t)DMODEL * DMODEL;

  // Transposed (N x K) bf16 weights for async LDS staging.
  bf* Wq_t  = (bf*)alloc(DD * 2);
  bf* Wk_t  = (bf*)alloc(DD * 2);
  bf* Wv_t  = (bf*)alloc(DD * 2);
  bf* Wrp_t = (bf*)alloc(DD * 2);
  bf* Wo_t  = (bf*)alloc(DD * 2);
  bf* gaT[6]; for (int i = 0; i < 6; ++i) gaT[i] = (bf*)alloc(DD * 2);
  bf* gfT[6]; for (int i = 0; i < 6; ++i) gfT[i] = (bf*)alloc(DD * 2);
  bf* ff1_t = (bf*)alloc((size_t)DMODEL * DFF_ * 2);   // DFF x D
  bf* ff2_t = (bf*)alloc((size_t)DFF_ * DMODEL * 2);   // D x DFF

  bf* x_bf  = (bf*)alloc((size_t)SB * DMODEL * 2);
  bf* kv_bf = (bf*)alloc((size_t)KB * DMODEL * 2);
  bf* rp_bf = (bf*)alloc((size_t)KLEN * DMODEL * 2);
  float* q_f = (float*)alloc((size_t)SB * DMODEL * 4);
  bf* qu_bf = (bf*)alloc((size_t)SB * DMODEL * 2);
  bf* qv_bf = (bf*)alloc((size_t)SB * DMODEL * 2);
  bf* k_bf  = (bf*)alloc((size_t)KB * DMODEL * 2);
  bf* v_bf  = (bf*)alloc((size_t)KB * DMODEL * 2);
  bf* r_bf  = (bf*)alloc((size_t)KLEN * DMODEL * 2);
  float* y_f = (float*)alloc((size_t)SB * DMODEL * 4);
  bf* y_bf  = (bf*)alloc((size_t)SB * DMODEL * 2);
  bf* yg_bf = (bf*)alloc((size_t)SB * DMODEL * 2);
  float* tA = (float*)alloc((size_t)SB * DMODEL * 4);
  float* tB = (float*)alloc((size_t)SB * DMODEL * 4);
  float* tC = (float*)alloc((size_t)SB * DMODEL * 4);
  float* tD = (float*)alloc((size_t)SB * DMODEL * 4);
  bf* rx_bf = (bf*)alloc((size_t)SB * DMODEL * 2);
  float* h_f = (float*)alloc((size_t)SB * DMODEL * 4);
  bf* h_bf  = (bf*)alloc((size_t)SB * DMODEL * 2);
  bf* hn_bf = (bf*)alloc((size_t)SB * DMODEL * 2);
  bf* ff1_o = (bf*)alloc((size_t)SB * DFF_ * 2);
  bf* ffr_bf = (bf*)alloc((size_t)SB * DMODEL * 2);

  auto ew_blocks = [](long long n) {
    long long b = (n + 1023) / 1024;
    if (b < 1) b = 1;
    if (b > 4096) b = 4096;
    return (int)b;
  };
  auto cvt = [&](const float* src, bf* dst, long long n) {
    cvt_kernel<<<ew_blocks(n), 256, 0, stream>>>(src, dst, n);
  };
  auto cvtT = [&](const float* src, bf* dst, int Kd, int N) {
    cvtT_kernel<<<ew_blocks((long long)Kd * N), 256, 0, stream>>>(src, dst, Kd, N);
  };
  auto gemm = [&](const bf* A, const bf* Wt, int M, int N, int Kd,
                  float* Cf, bf* Cb, const float* bias, int flags) {
    dim3 grid(M / GBM, N / GBN);
    gemm_kernel<<<grid, 256, 0, stream>>>(A, Wt, M, N, Kd, Cf, Cb, bias, flags);
  };
  const long long nSD = (long long)SB * DMODEL;

  // ---- Weight conversion to transposed bf16, input conversion ----
  cvtT(Wq, Wq_t, DMODEL, DMODEL);   cvtT(Wk, Wk_t, DMODEL, DMODEL);
  cvtT(Wv, Wv_t, DMODEL, DMODEL);   cvtT(Wrp, Wrp_t, DMODEL, DMODEL);
  cvtT(Wo, Wo_t, DMODEL, DMODEL);
  for (int i = 0; i < 6; ++i) cvtT(gaW[i], gaT[i], DMODEL, DMODEL);
  for (int i = 0; i < 6; ++i) cvtT(gfW[i], gfT[i], DMODEL, DMODEL);
  cvtT(ff_w1, ff1_t, DMODEL, DFF_);
  cvtT(ff_w2, ff2_t, DFF_, DMODEL);
  cvt(x, x_bf, nSD);
  cvt(relpos, rp_bf, (long long)KLEN * DMODEL);

  // ---- LN: kv = [ln(memory); ln(x)] (flattened (K,B,D) row order) ----
  ln_kernel<<<SB, 256, 0, stream>>>(memory, ln1_g, ln1_b, kv_bf);
  ln_kernel<<<SB, 256, 0, stream>>>(x, ln1_g, ln1_b, kv_bf + (size_t)SB * DMODEL);
  const bf* xnorm_bf = kv_bf + (size_t)SB * DMODEL;

  // ---- Projections ----
  gemm(xnorm_bf, Wq_t, SB, DMODEL, DMODEL, q_f, nullptr, nullptr, 0);
  qbias_kernel<<<ew_blocks(nSD), 256, 0, stream>>>(q_f, u_bias, v_bias,
                                                   qu_bf, qv_bf, nSD);
  gemm(kv_bf, Wk_t, KB, DMODEL, DMODEL, nullptr, k_bf, nullptr, 0);
  gemm(kv_bf, Wv_t, KB, DMODEL, DMODEL, nullptr, v_bf, nullptr, 0);
  gemm(rp_bf, Wrp_t, KLEN, DMODEL, DMODEL, nullptr, r_bf, nullptr, 0);

  // ---- Attention (136 KB dynamic LDS; CDNA5 supports 320 KB/workgroup) ----
  const int ATTN_LDS = 8192 + 65536 + 65536;
  hipFuncSetAttribute((const void*)attn_kernel,
                      hipFuncAttributeMaxDynamicSharedMemorySize, ATTN_LDS);
  attn_kernel<<<dim3(S_LEN / 32, NHEAD, BATCH), 256, ATTN_LDS, stream>>>(
      qu_bf, qv_bf, k_bf, v_bf, r_bf, y_f);

  // ---- Output projection + GRU 1 (x = input x, y = relu(attn_out)) ----
  cvt(y_f, y_bf, nSD);
  gemm(y_bf, Wo_t, SB, DMODEL, DMODEL, tA, nullptr, nullptr, 0);
  relu_bf_kernel<<<ew_blocks(nSD), 256, 0, stream>>>(tA, yg_bf, nSD);

  gemm(yg_bf, gaT[0], SB, DMODEL, DMODEL, tB, nullptr, nullptr, 0);  // y@Wr
  gemm(x_bf,  gaT[1], SB, DMODEL, DMODEL, tB, nullptr, nullptr, 1);  // +x@Ur
  sigmoid_kernel<<<ew_blocks(nSD), 256, 0, stream>>>(tB, nullptr, nSD);
  gemm(yg_bf, gaT[2], SB, DMODEL, DMODEL, tC, nullptr, nullptr, 0);  // y@Wz
  gemm(x_bf,  gaT[3], SB, DMODEL, DMODEL, tC, nullptr, nullptr, 1);  // +x@Uz
  sigmoid_kernel<<<ew_blocks(nSD), 256, 0, stream>>>(tC, ga_bias, nSD);
  mul_bf_kernel<<<ew_blocks(nSD), 256, 0, stream>>>(tB, x, rx_bf, nSD);
  gemm(yg_bf, gaT[4], SB, DMODEL, DMODEL, tD, nullptr, nullptr, 0);  // y@Wg
  gemm(rx_bf, gaT[5], SB, DMODEL, DMODEL, tD, nullptr, nullptr, 1);  // +(r*x)@Ug
  tanh_kernel<<<ew_blocks(nSD), 256, 0, stream>>>(tD, nSD);
  gru_out_kernel<<<ew_blocks(nSD), 256, 0, stream>>>(tC, x, tD, h_f, nSD);

  // ---- FFN ----
  ln_kernel<<<SB, 256, 0, stream>>>(h_f, ln2_g, ln2_b, hn_bf);
  gemm(hn_bf, ff1_t, SB, DFF_, DMODEL, nullptr, ff1_o, ff_b1, 2 | 4);  // bias+relu
  gemm(ff1_o, ff2_t, SB, DMODEL, DFF_, tA, nullptr, ff_b2, 2);         // bias
  relu_bf_kernel<<<ew_blocks(nSD), 256, 0, stream>>>(tA, ffr_bf, nSD);
  cvt(h_f, h_bf, nSD);

  // ---- GRU 2 (x = h, y = relu(ff)), writes final output ----
  gemm(ffr_bf, gfT[0], SB, DMODEL, DMODEL, tB, nullptr, nullptr, 0);
  gemm(h_bf,   gfT[1], SB, DMODEL, DMODEL, tB, nullptr, nullptr, 1);
  sigmoid_kernel<<<ew_blocks(nSD), 256, 0, stream>>>(tB, nullptr, nSD);
  gemm(ffr_bf, gfT[2], SB, DMODEL, DMODEL, tC, nullptr, nullptr, 0);
  gemm(h_bf,   gfT[3], SB, DMODEL, DMODEL, tC, nullptr, nullptr, 1);
  sigmoid_kernel<<<ew_blocks(nSD), 256, 0, stream>>>(tC, gf_bias, nSD);
  mul_bf_kernel<<<ew_blocks(nSD), 256, 0, stream>>>(tB, h_f, rx_bf, nSD);
  gemm(ffr_bf, gfT[4], SB, DMODEL, DMODEL, tD, nullptr, nullptr, 0);
  gemm(rx_bf,  gfT[5], SB, DMODEL, DMODEL, tD, nullptr, nullptr, 1);
  tanh_kernel<<<ew_blocks(nSD), 256, 0, stream>>>(tD, nSD);
  gru_out_kernel<<<ew_blocks(nSD), 256, 0, stream>>>(tC, h_f, tD,
                                                     (float*)d_out, nSD);
}
